// HopfieldLayer_46471546143152
// MI455X (gfx1250) — matrix operand order repaired and checked by reading, not compiled, not run on hardware
//
#include <hip/hip_runtime.h>
#include <math.h>

#define DD     2048
#define BBATCH 256
#define NELEM  (BBATCH * DD)      // 524288
#define WELEM  (DD * DD)          // 4194304
#define DT_F   0.1f
#define TOL_F  0.01f
#define NSTEPS 50
#define EPS_F  1e-8f

typedef __attribute__((ext_vector_type(16))) __bf16    v16bf;
typedef __attribute__((ext_vector_type(8)))  float     v8f;
typedef __attribute__((ext_vector_type(4)))  unsigned  u32x4t;
typedef __attribute__((ext_vector_type(4)))  int       i32x4t;
typedef __attribute__((ext_vector_type(8)))  int       i32x8t;

union FragU {
    v16bf  v;
    u32x4t q[2];
};

// ---------------------------------------------------------------------------
// TDM: DMA a 2-D tile (rows x 32 bf16 elements, row stride 2048 elements)
// from global memory into LDS at byte offset lds_off.  Descriptor per ISA
// CDNA5 ISA:
//   group0: count=1 | lds_addr | global_addr[56:0] | type=2
//   group1: data_size=1(2B), tensor_dim0=2048, tensor_dim1=2048,
//           tile_dim0=32, tile_dim1=rows, tensor_dim0_stride=2048
//   groups 2/3: zero (2-D tile, tile_dim2=0)
// Tracked with TENSORcnt; completion enforced with s_wait_tensorcnt.
// ---------------------------------------------------------------------------
__device__ __forceinline__ void tdm_load_tile(const void* gptr, unsigned lds_off,
                                              unsigned rows)
{
    const unsigned long long ga = (unsigned long long)(uintptr_t)gptr;
    u32x4t g0 = { 1u,                                   // count=1, user mode
                  lds_off,                              // lds_addr (bytes)
                  (unsigned)ga,                         // global_addr[31:0]
                  ((unsigned)((ga >> 32) & 0x01FFFFFFull)) | 0x80000000u }; // [56:32] | type=2
    i32x8t g1 = { (int)0x00010000u,                     // data_size=1 -> 2 bytes/elem
                  (int)(2048u << 16),                   // tensor_dim0[15:0] in [31:16]
                  (int)(2048u << 16),                   // dim0 hi=0 | tensor_dim1[15:0]
                  (int)(32u << 16),                     // dim1 hi=0 | tile_dim0=32
                  (int)rows,                            // tile_dim1=rows | tile_dim2=0
                  (int)2048,                            // tensor_dim0_stride low 32
                  0, 0 };                               // stride hi / dim1_stride = 0
    i32x4t z4 = { 0, 0, 0, 0 };
#if __has_include(<hip/amd_detail/amd_gfx1250_TDM.h>)
    i32x8t z8 = { 0, 0, 0, 0, 0, 0, 0, 0 };
    __builtin_amdgcn_tensor_load_to_lds(g0, g1, z4, z4, z8, 0);
#else
    __builtin_amdgcn_tensor_load_to_lds(g0, g1, z4, z4, 0);
#endif
}

// ---------------------------------------------------------------------------
// WMMA GEMM: C[256 x 2048] = A[256 x 2048](bf16, row-major)
//                          x B[2048 x 2048], B supplied N-major (Bt[n][k]).
// Block: 8 waves -> 32 rows x 256 cols of C (wave w: m-tile w&1, n-group w>>1).
// Per K-chunk of 32, wave 0 TDM-stages A(32x32) and B(256x32) panels into
// double-buffered LDS (64 B per staged row); waves consume via ds_load_b128.
// MODE 0: epilogue h = tanh(z + bias), store H f32, atomic per-row sum(h^2).
// MODE 1: plain f32 store.
// ---------------------------------------------------------------------------
template <int MODE>
__global__ __launch_bounds__(256)
void wmma_gemm(const __bf16* __restrict__ A, const __bf16* __restrict__ Bt,
               const float* __restrict__ bias, float* __restrict__ out,
               float* __restrict__ rowsum)
{
    __shared__ __align__(64) unsigned char ldsA[2][32 * 64];    //  2 x 2 KB
    __shared__ __align__(64) unsigned char ldsB[2][256 * 64];   //  2 x 16 KB

    const int lane  = threadIdx.x & 31;
    const int w     = threadIdx.x >> 5;   // wave in block 0..7
    const int mblk  = blockIdx.x & 7;     // 8 row blocks of 32
    const int nblk  = blockIdx.x >> 3;    // 8 col blocks of 256
    const int mt    = w & 1;              // 16-row tile within block
    const int ng    = w >> 1;             // 64-col group within block
    const int laneN = lane & 15;
    const bool hi   = lane >= 16;

    const unsigned aoff[2] = { (unsigned)(uintptr_t)&ldsA[0][0],
                               (unsigned)(uintptr_t)&ldsA[1][0] };
    const unsigned boff[2] = { (unsigned)(uintptr_t)&ldsB[0][0],
                               (unsigned)(uintptr_t)&ldsB[1][0] };

    const __bf16* Abase = A  + (size_t)(mblk * 32)  * DD;
    const __bf16* Bbase = Bt + (size_t)(nblk * 256) * DD;

    if (w == 0) {                         // prologue: stage buffer 0 (k = 0)
        tdm_load_tile(Abase, aoff[0], 32);
        tdm_load_tile(Bbase, boff[0], 256);
    }

    v8f acc[4] = {};

    for (int k = 0; k < DD; k += 32) {
        const int p = (k >> 5) & 1;
        if (w == 0) {
            if (k + 32 < DD) {            // stage next chunk into buffer p^1
                tdm_load_tile(Abase + k + 32, aoff[p ^ 1], 32);
                tdm_load_tile(Bbase + k + 32, boff[p ^ 1], 256);
                __builtin_amdgcn_s_wait_tensorcnt(2);  // oldest pair (buf p) done
            } else {
                __builtin_amdgcn_s_wait_tensorcnt(0);
            }
        }
        __syncthreads();                  // buffer p published to all waves

        // A fragment (ISA 16-bit A 16x32 layout): row mt*16+laneN,
        // byte chunks (hi?16:0) and +32 within the 64 B staged row.
        FragU a;
        const unsigned char* pa = &ldsA[p][(mt * 16 + laneN) * 64 + (hi ? 16 : 0)];
        a.q[0] = *(const u32x4t*)(pa);
        a.q[1] = *(const u32x4t*)(pa + 32);

#pragma unroll
        for (int t = 0; t < 4; ++t) {
            // B fragment: Bt row ng*64+t*16+laneN, byte chunks (hi?16:0) and +32.
            const unsigned char* pb =
                &ldsB[p][(ng * 64 + t * 16 + laneN) * 64 + (hi ? 16 : 0)];
            FragU b;
            b.q[0] = *(const u32x4t*)(pb);
            b.q[1] = *(const u32x4t*)(pb + 32);
            acc[t] = __builtin_amdgcn_wmma_f32_16x16x32_bf16(
                false, a.v, false, b.v, (short)0, acc[t], false, false);
        }
        __syncthreads();                  // all waves done with buffer p
    }

    // C layout: VGPR r, lanes 0-15 -> M=r, lanes 16-31 -> M=8+r; N = lane%16.
#pragma unroll
    for (int r = 0; r < 8; ++r) {
        const int row = mblk * 32 + mt * 16 + r + (hi ? 8 : 0);
        if (MODE == 0) {
            float s = 0.f;
#pragma unroll
            for (int t = 0; t < 4; ++t) {
                const int col = nblk * 256 + ng * 64 + t * 16 + laneN;
                const float h = tanhf(acc[t][r] + bias[col]);
                out[(size_t)row * DD + col] = h;
                s += h * h;
            }
            s += __shfl_xor(s, 1, 16);
            s += __shfl_xor(s, 2, 16);
            s += __shfl_xor(s, 4, 16);
            s += __shfl_xor(s, 8, 16);
            if (laneN == 0) atomicAdd(&rowsum[row], s);
        } else {
#pragma unroll
            for (int t = 0; t < 4; ++t) {
                const int col = nblk * 256 + ng * 64 + t * 16 + laneN;
                out[(size_t)row * DD + col] = acc[t][r];
            }
        }
    }
}

// ---------------------------------------------------------------------------
// Elementwise / setup kernels
// ---------------------------------------------------------------------------
__global__ void prep_w(const float* __restrict__ W, __bf16* __restrict__ Wb,
                       __bf16* __restrict__ WTb)
{
    const int i = blockIdx.x * blockDim.x + threadIdx.x;
    if (i >= WELEM) return;
    const int r = i >> 11, c = i & (DD - 1);
    const float w = W[i];
    Wb[i] = (__bf16)w;
    WTb[(size_t)c * DD + r] = (__bf16)w;
}

__global__ void init_state(const float* __restrict__ x0, float* __restrict__ xcur,
                           unsigned* __restrict__ maxbits, int* __restrict__ done)
{
    const int i = blockIdx.x * blockDim.x + threadIdx.x;
    if (i < NELEM) xcur[i] = x0[i];
    if (i == 0) { *maxbits = 0u; *done = 0; }
}

__global__ void conv_x(const float* __restrict__ xe, __bf16* __restrict__ Ab,
                       float* __restrict__ rowsum)
{
    const int i = blockIdx.x * blockDim.x + threadIdx.x;
    if (i >= NELEM) return;
    Ab[i] = (__bf16)xe[i];
    if (i < BBATCH) rowsum[i] = 0.f;
}

__global__ void gvec(const float* __restrict__ H, const float* __restrict__ rowsum,
                     __bf16* __restrict__ Ab)
{
    const int i = blockIdx.x * blockDim.x + threadIdx.x;
    if (i >= NELEM) return;
    const int m = i >> 11;
    const float h = H[i];
    const float y = sqrtf(rowsum[m]);
    const float coef = 5.f / (y + EPS_F);
    Ab[i] = (__bf16)(-coef * h * (1.f - h * h));
}

__global__ void reduce_absmax(const float* __restrict__ k, unsigned* __restrict__ maxbits)
{
    __shared__ unsigned smax[256];
    unsigned m = 0u;
    for (int i = blockIdx.x * blockDim.x + threadIdx.x; i < NELEM;
         i += gridDim.x * blockDim.x) {
        const unsigned v = __float_as_uint(fabsf(k[i]));
        m = v > m ? v : m;
    }
    smax[threadIdx.x] = m;
    __syncthreads();
    for (int s = 128; s > 0; s >>= 1) {
        if ((int)threadIdx.x < s) {
            const unsigned v = smax[threadIdx.x + s];
            if (v > smax[threadIdx.x]) smax[threadIdx.x] = v;
        }
        __syncthreads();
    }
    if (threadIdx.x == 0) atomicMax(maxbits, smax[0]);
}

__global__ void done_update(unsigned* __restrict__ maxbits, int* __restrict__ done)
{
    if (__uint_as_float(*maxbits) < TOL_F) *done = 1;
    *maxbits = 0u;
}

__global__ void rk_stage(const float* __restrict__ x, const float* __restrict__ k,
                         float* __restrict__ accum, float* __restrict__ xe,
                         float aAcc, float aEval, int first)
{
    const int i = blockIdx.x * blockDim.x + threadIdx.x;
    if (i >= NELEM) return;
    const float kv = k[i];
    const float base = (first ? x[i] : accum[i]) + aAcc * kv;
    accum[i] = base;
    if (aEval != 0.f) xe[i] = x[i] + aEval * kv;
}

__global__ void rk_finish(float* __restrict__ xcur, const float* __restrict__ accum,
                          const int* __restrict__ done)
{
    const int i = blockIdx.x * blockDim.x + threadIdx.x;
    if (i >= NELEM) return;
    if (!*done) xcur[i] = accum[i];
}

__global__ void copy_out(const float* __restrict__ xcur, float* __restrict__ out)
{
    const int i = blockIdx.x * blockDim.x + threadIdx.x;
    if (i < NELEM) out[i] = xcur[i];
}

// ---------------------------------------------------------------------------
// Host driver: 50 RK4 steps, 4 pvf evals each, all on `stream`.
// ---------------------------------------------------------------------------
extern "C" void kernel_launch(void* const* d_in, const int* in_sizes, int n_in,
                              void* d_out, int out_size, void* d_ws, size_t ws_size,
                              hipStream_t stream)
{
    (void)in_sizes; (void)n_in; (void)out_size; (void)ws_size;
    const float* x0   = (const float*)d_in[0];
    const float* W    = (const float*)d_in[1];
    const float* bias = (const float*)d_in[2];

    char* ws = (char*)d_ws;
    size_t off = 0;
    auto alloc = [&](size_t bytes) -> char* {
        char* p = ws + off;
        off = (off + bytes + 255) & ~(size_t)255;
        return p;
    };
    __bf16*   Wb      = (__bf16*)alloc((size_t)WELEM * 2);
    __bf16*   WTb     = (__bf16*)alloc((size_t)WELEM * 2);
    float*    xcur    = (float*)alloc((size_t)NELEM * 4);
    float*    xe      = (float*)alloc((size_t)NELEM * 4);
    float*    accb    = (float*)alloc((size_t)NELEM * 4);
    float*    kb      = (float*)alloc((size_t)NELEM * 4);
    float*    Hb      = (float*)alloc((size_t)NELEM * 4);
    __bf16*   Ab      = (__bf16*)alloc((size_t)NELEM * 2);
    float*    rowsum  = (float*)alloc((size_t)BBATCH * 4);
    unsigned* maxbits = (unsigned*)alloc(256);
    int*      done    = (int*)alloc(256);

    const dim3 blk(256);
    const dim3 gE((NELEM + 255) / 256);
    const dim3 gW((WELEM + 255) / 256);
    const dim3 gG(64);   // 8 row-blocks x 8 col-blocks, 8 waves each

    prep_w<<<gW, blk, 0, stream>>>(W, Wb, WTb);
    init_state<<<gE, blk, 0, stream>>>(x0, xcur, maxbits, done);

    auto pvf = [&](const float* xin, float* kout) {
        conv_x<<<gE, blk, 0, stream>>>(xin, Ab, rowsum);
        wmma_gemm<0><<<gG, blk, 0, stream>>>(Ab, WTb, bias, Hb, rowsum);
        gvec<<<gE, blk, 0, stream>>>(Hb, rowsum, Ab);
        wmma_gemm<1><<<gG, blk, 0, stream>>>(Ab, Wb, bias, kout, rowsum);
    };

    for (int s = 0; s < NSTEPS; ++s) {
        pvf(xcur, kb);                                                      // k1
        reduce_absmax<<<dim3(256), blk, 0, stream>>>(kb, maxbits);
        done_update<<<1, 1, 0, stream>>>(maxbits, done);
        rk_stage<<<gE, blk, 0, stream>>>(xcur, kb, accb, xe,
                                         DT_F / 6.f, 0.5f * DT_F, 1);
        pvf(xe, kb);                                                        // k2
        rk_stage<<<gE, blk, 0, stream>>>(xcur, kb, accb, xe,
                                         DT_F / 3.f, 0.5f * DT_F, 0);
        pvf(xe, kb);                                                        // k3
        rk_stage<<<gE, blk, 0, stream>>>(xcur, kb, accb, xe,
                                         DT_F / 3.f, DT_F, 0);
        pvf(xe, kb);                                                        // k4
        rk_stage<<<gE, blk, 0, stream>>>(xcur, kb, accb, xe,
                                         DT_F / 6.f, 0.f, 0);
        rk_finish<<<gE, blk, 0, stream>>>(xcur, accb, done);
    }
    copy_out<<<gE, blk, 0, stream>>>(xcur, (float*)d_out);
}
